// ImageGPTAttention_55336358641971
// MI455X (gfx1250) — matrix operand order repaired and checked by reading, not compiled, not run on hardware
//
#include <hip/hip_runtime.h>
#include <hip/hip_bf16.h>

// ---------------------------------------------------------------------------
// GPT-2 attention block on gfx1250 (MI455X). All matmuls via
// v_wmma_f32_16x16x32_bf16 (wave32 WMMA, f32 accumulate).
//   Kernel 1: QKV GEMM  [8192,1024]f32 @ [1024,3072]f32 -> bf16 qkv (ws)
//   Kernel 2: fused causal flash attention, async-LDS K staging -> bf16 (ws)
//   Kernel 3: proj GEMM [8192,1024]bf16 @ [1024,1024]f32 + bias -> f32 out
// GEMM: 128x128 block tile, 32x64 wave tile (8 WMMA per K-step per wave),
// double-buffered LDS + register-pipelined globals, one barrier per K-tile,
// L2 prefetch of weights two tiles ahead.
// ---------------------------------------------------------------------------

typedef __bf16 bf16;
typedef __attribute__((ext_vector_type(16))) __bf16 v16bf;
typedef __attribute__((ext_vector_type(8)))  __bf16 v8bf;
typedef __attribute__((ext_vector_type(8)))  float  v8f;

#define WMMA_BF16(a, b, c) \
  __builtin_amdgcn_wmma_f32_16x16x32_bf16(false, (a), false, (b), (short)0, (c), false, false)

// Async global->LDS copy, 16B per lane, tracked by ASYNCcnt (CDNA5 path).
__device__ __forceinline__ void async_load_b128(unsigned lds_off, const void* g) {
  asm volatile("global_load_async_to_lds_b128 %0, %1, off"
               :: "v"(lds_off), "v"((unsigned long long)(uintptr_t)g)
               : "memory");
}
__device__ __forceinline__ void wait_asynccnt0() {
  asm volatile("s_wait_asynccnt 0x0" ::: "memory");
}

// xor-shuffle within 16-lane rows via DPP ROW_XMASK (keeps the DS pipe free).
template <int XMASK>
__device__ __forceinline__ float dpp_xor(float x) {
  int v = __builtin_amdgcn_update_dpp(0, __float_as_int(x),
                                      0x160 | XMASK, 0xf, 0xf, true);
  return __int_as_float(v);
}
__device__ __forceinline__ float row_max16(float x) {
  x = fmaxf(x, dpp_xor<1>(x));
  x = fmaxf(x, dpp_xor<2>(x));
  x = fmaxf(x, dpp_xor<4>(x));
  x = fmaxf(x, dpp_xor<8>(x));
  return x;
}
__device__ __forceinline__ float row_sum16(float x) {
  x += dpp_xor<1>(x);
  x += dpp_xor<2>(x);
  x += dpp_xor<4>(x);
  x += dpp_xor<8>(x);
  return x;
}

// A-fragment (16x32 bf16 MxK): lane m<16 -> M=m, K = hi*16 + j.
// p = &tile[(rowbase + (lane&15))*stride + kbase + (lane>>4)*16]
__device__ __forceinline__ v16bf ld_frag_a(const bf16* p) {
  v8bf lo = *(const v8bf*)(p);
  v8bf hh = *(const v8bf*)(p + 8);
  v16bf f;
#pragma unroll
  for (int j = 0; j < 8; ++j) { f[j] = lo[j]; f[j + 8] = hh[j]; }
  return f;
}
// B-fragment (32x16 bf16 KxN) from a TRANSPOSED tile Bt[N][K]:
// lane n<16 -> N=n, K = hi*16 + j, contiguous in Bt row n.
// p = &Bt[(colbase + (lane&15))*stride + kbase + (lane>>4)*16]
__device__ __forceinline__ v16bf ld_frag_b(const bf16* p) {
  v8bf lo = *(const v8bf*)(p);
  v8bf hh = *(const v8bf*)(p + 8);
  v16bf f;
#pragma unroll
  for (int j = 0; j < 8; ++j) { f[j] = lo[j]; f[j + 8] = hh[j]; }
  return f;
}

// ---------------------------------------------------------------------------
// GEMM: C[M,N] = A[M,K] @ W[K,N] + bias[N].  128x128 block tile, BK=32,
// 256 threads = 8 waves in a 4x2 grid, each wave a 32x64 subtile
// (2 A-frags x 4 B-frags -> 8 back-to-back WMMAs per K-step).
// ---------------------------------------------------------------------------
template <bool A_F32, bool OUT_BF16>
__global__ __launch_bounds__(256) void wmma_gemm_kernel(
    const void* __restrict__ Av, const float* __restrict__ W,
    const float* __restrict__ bias, void* __restrict__ Cv,
    int M, int N, int K) {
  constexpr int BM = 128, BN = 128, BK = 32;
  constexpr int LDA = 48, LDB = 48;   // 96B rows: 16B aligned, non-pow2
  __shared__ bf16 As[2][BM * LDA];
  __shared__ bf16 Bt[2][BN * LDB];    // W tile transposed: [N=128][K=32]

  const int tid  = threadIdx.x;
  const int lane = tid & 31;
  const int wave = tid >> 5;
  const int wm = wave >> 1;           // 0..3 -> 32-row stripe
  const int wn = wave & 1;            // 0..1 -> 64-col stripe
  const int m0 = blockIdx.y * BM;
  const int n0 = blockIdx.x * BN;
  const int mlane = lane & 15, hilane = lane >> 4;
  (void)M;

  const v8f vzero = {0.f, 0.f, 0.f, 0.f, 0.f, 0.f, 0.f, 0.f};
  v8f acc[2][4];
#pragma unroll
  for (int i = 0; i < 2; ++i)
#pragma unroll
    for (int j = 0; j < 4; ++j) acc[i][j] = vzero;

  const int arow = tid >> 1;          // 0..127
  const int acol = (tid & 1) * 16;    // 0,16

  float4 aregf[4];   // staged A chunk, f32 path (16 f32)
  v8bf   aregb[2];   // staged A chunk, bf16 path (16 bf16)
  float4 wreg[4];    // staged W chunk (16 f32)

  auto gload = [&](int kt) {
    if (A_F32) {
      const float* A = (const float*)Av;
      const float4* g = (const float4*)(A + (size_t)(m0 + arow) * K + kt + acol);
#pragma unroll
      for (int i = 0; i < 4; ++i) aregf[i] = g[i];
    } else {
      const bf16* A = (const bf16*)Av;
      const v8bf* g = (const v8bf*)(A + (size_t)(m0 + arow) * K + kt + acol);
      aregb[0] = g[0];
      aregb[1] = g[1];
    }
#pragma unroll
    for (int i = 0; i < 4; ++i) {
      int fid = tid + i * 256;
      int wr = fid >> 5, wc = (fid & 31) * 4;
      wreg[i] = *(const float4*)(W + (size_t)(kt + wr) * N + n0 + wc);
    }
    // warm L2 for the weight stream two K-tiles ahead (global_prefetch_b8)
    if (kt + 2 * BK < K)
      __builtin_prefetch(W + (size_t)(kt + 2 * BK + (tid >> 5)) * N + n0 + (tid & 31) * 4, 0, 1);
  };
  auto lstore = [&](int buf) {
    if (A_F32) {
      v8bf c0, c1;
#pragma unroll
      for (int i = 0; i < 2; ++i) {
        c0[i * 4 + 0] = (bf16)aregf[i].x;     c0[i * 4 + 1] = (bf16)aregf[i].y;
        c0[i * 4 + 2] = (bf16)aregf[i].z;     c0[i * 4 + 3] = (bf16)aregf[i].w;
        c1[i * 4 + 0] = (bf16)aregf[i + 2].x; c1[i * 4 + 1] = (bf16)aregf[i + 2].y;
        c1[i * 4 + 2] = (bf16)aregf[i + 2].z; c1[i * 4 + 3] = (bf16)aregf[i + 2].w;
      }
      *(v8bf*)&As[buf][arow * LDA + acol]     = c0;
      *(v8bf*)&As[buf][arow * LDA + acol + 8] = c1;
    } else {
      *(v8bf*)&As[buf][arow * LDA + acol]     = aregb[0];
      *(v8bf*)&As[buf][arow * LDA + acol + 8] = aregb[1];
    }
#pragma unroll
    for (int i = 0; i < 4; ++i) {
      int fid = tid + i * 256;
      int wr = fid >> 5, wc = (fid & 31) * 4;
      Bt[buf][(wc + 0) * LDB + wr] = (bf16)wreg[i].x;
      Bt[buf][(wc + 1) * LDB + wr] = (bf16)wreg[i].y;
      Bt[buf][(wc + 2) * LDB + wr] = (bf16)wreg[i].z;
      Bt[buf][(wc + 3) * LDB + wr] = (bf16)wreg[i].w;
    }
  };

  gload(0);
  int cur = 0;
  const int ntiles = K / BK;
  for (int t = 0; t < ntiles; ++t) {
    lstore(cur);
    __syncthreads();                          // cur visible; fences t-1 readers
    if (t + 1 < ntiles) gload((t + 1) * BK);  // in flight during WMMAs below

    const bf16* asb = &As[cur][0];
    const bf16* btb = &Bt[cur][0];
    v16bf a[2], b[4];
#pragma unroll
    for (int i = 0; i < 2; ++i)
      a[i] = ld_frag_a(&asb[(wm * 32 + i * 16 + mlane) * LDA + hilane * 16]);
#pragma unroll
    for (int j = 0; j < 4; ++j)
      b[j] = ld_frag_b(&btb[(wn * 64 + j * 16 + mlane) * LDB + hilane * 16]);
#pragma unroll
    for (int i = 0; i < 2; ++i)
#pragma unroll
      for (int j = 0; j < 4; ++j)
        acc[i][j] = WMMA_BF16(a[i], b[j], acc[i][j]);
    cur ^= 1;
  }

  // C-fragment layout: col = lane&15, row = r + 8*(lane>>4)
#pragma unroll
  for (int i = 0; i < 2; ++i)
#pragma unroll
    for (int j = 0; j < 4; ++j) {
      int col = n0 + wn * 64 + j * 16 + mlane;
      float bv = bias[col];
#pragma unroll
      for (int r = 0; r < 8; ++r) {
        int row = m0 + wm * 32 + i * 16 + hilane * 8 + r;
        float v = acc[i][j][r] + bv;
        if (OUT_BF16) ((bf16*)Cv)[(size_t)row * N + col] = (bf16)v;
        else          ((float*)Cv)[(size_t)row * N + col] = v;
      }
    }
}

// ---------------------------------------------------------------------------
// Fused causal flash attention.
// qkv: [B*S, 3E] bf16 (Q | K | V), out: [B*S, E] bf16.
// grid = (S/64, B*H), 128 threads = 4 waves, wave w owns q-rows q0+16w..+15.
// K tiles staged via global_load_async_to_lds_b128 (ASYNCcnt), V tiles via
// register-pipelined loads + transposed DS stores; both double-buffered.
// ---------------------------------------------------------------------------
__global__ __launch_bounds__(128) void flash_attn_kernel(
    const bf16* __restrict__ qkv, bf16* __restrict__ attn_out) {
  constexpr int S = 2048, E = 1024, Dh = 64, NE3 = 3072;
  constexpr int LDK = 72;   // 144B rows (16B-aligned, non-pow2)
  constexpr int LDV = 48;
  constexpr int LDP = 48;
  __shared__ bf16 Ks[2][32 * LDK];      // [key=32][d=64]  (native B-layout)
  __shared__ bf16 Vt[2][Dh * LDV];      // [d=64][key=32]  (transposed V)
  __shared__ bf16 Ps[4 * 16 * LDP];     // per-wave 16x32 probability tile

  const int tid  = threadIdx.x;
  const int lane = tid & 31;
  const int wave = tid >> 5;
  const int mlane = lane & 15, hilane = lane >> 4;

  const int qt = blockIdx.x;
  const int bh = blockIdx.y;
  const int bb = bh >> 4, h = bh & 15;
  const int q0 = qt * 64;
  const size_t rowbase = (size_t)bb * S;

  // staging geometry: thread t covers key-row tid>>2, 16 d-cols (tid&3)*16
  const int vr = tid >> 2;
  const int vc = (tid & 3) * 16;

  auto issueK = [&](int k0, int buf) {   // async global->LDS, no transpose
    const bf16* gk = qkv + (rowbase + k0 + vr) * (size_t)NE3 + E + h * Dh + vc;
    unsigned l0 = (unsigned)(uintptr_t)&Ks[buf][vr * LDK + vc];
    async_load_b128(l0,      gk);
    async_load_b128(l0 + 16, gk + 8);
  };
  v8bf vreg0, vreg1;
  auto gloadV = [&](int k0) {
    const bf16* gv = qkv + (rowbase + k0 + vr) * (size_t)NE3 + 2 * E + h * Dh + vc;
    vreg0 = *(const v8bf*)gv;
    vreg1 = *(const v8bf*)(gv + 8);
  };
  auto storeV = [&](int buf) {           // transpose into Vt
    bf16* vt = &Vt[buf][0];
#pragma unroll
    for (int j = 0; j < 8; ++j) {
      vt[(vc + j) * LDV + vr]     = vreg0[j];
      vt[(vc + 8 + j) * LDV + vr] = vreg1[j];
    }
  };

  issueK(0, 0);       // prefetch first K tile (async)
  gloadV(0);          // first V tile into registers

  // Q fragments, held in registers for the whole block
  v16bf qa[2];
  {
    const bf16* qp = qkv + (rowbase + q0 + wave * 16 + mlane) * (size_t)NE3 + h * Dh;
#pragma unroll
    for (int t = 0; t < 2; ++t)
      qa[t] = ld_frag_a(qp + t * 32 + hilane * 16);
  }

  const v8f vzero = {0.f, 0.f, 0.f, 0.f, 0.f, 0.f, 0.f, 0.f};
  v8f o[4];
#pragma unroll
  for (int t = 0; t < 4; ++t) o[t] = vzero;
  float mrow[8], lrow[8];
#pragma unroll
  for (int r = 0; r < 8; ++r) { mrow[r] = -1e30f; lrow[r] = 0.f; }

  int cur = 0;
  const int nkt = qt * 2 + 2;   // causal: key tiles up to q0+63
  for (int kt = 0; kt < nkt; ++kt) {
    const int k0 = kt * 32;
    storeV(cur);          // DS stores; visible after the barrier below
    wait_asynccnt0();     // my K(cur) async copies have landed in LDS
    __syncthreads();      // everyone's K/V for cur staged; fences kt-1 readers
    if (kt + 1 < nkt) {   // prefetch next tile during this tile's WMMAs
      issueK(k0 + 32, cur ^ 1);
      gloadV(k0 + 32);
    }

    const bf16* kb = &Ks[cur][0];
    const bf16* vtb = &Vt[cur][0];

    // ---- scores S[16x32] = Q(16x64) x K^T(64x32) ----
    v8f s0 = vzero, s1 = vzero;
#pragma unroll
    for (int t = 0; t < 2; ++t) {
      v16bf bk0 = ld_frag_b(&kb[(0  + mlane) * LDK + t * 32 + hilane * 16]);
      v16bf bk1 = ld_frag_b(&kb[(16 + mlane) * LDK + t * 32 + hilane * 16]);
      s0 = WMMA_BF16(qa[t], bk0, s0);
      s1 = WMMA_BF16(qa[t], bk1, s1);
    }

    // ---- online softmax (f32, DPP row reductions) ----
    const float scale = 0.125f;  // 1/sqrt(64)
    const int rbase = q0 + wave * 16 + hilane * 8;
    const int c0 = k0 + mlane, c1 = c0 + 16;
    float e0v[8], e1v[8];
#pragma unroll
    for (int r = 0; r < 8; ++r) {
      int row = rbase + r;
      float a0 = (c0 <= row) ? s0[r] * scale : -1e30f;
      float a1 = (c1 <= row) ? s1[r] * scale : -1e30f;
      float mx = row_max16(fmaxf(a0, a1));
      float mnew = fmaxf(mrow[r], mx);           // mnew >= mrow
      float corr = __expf(mrow[r] - mnew);       // in (0,1]
      float e0 = (c0 <= row) ? __expf(a0 - mnew) : 0.f;
      float e1 = (c1 <= row) ? __expf(a1 - mnew) : 0.f;
      float rs = row_sum16(e0 + e1);
      lrow[r] = lrow[r] * corr + rs;
      mrow[r] = mnew;
#pragma unroll
      for (int t = 0; t < 4; ++t) o[t][r] *= corr;
      e0v[r] = e0; e1v[r] = e1;
    }

    // ---- re-layout P: C-fragment -> per-wave LDS tile -> A-fragment ----
    bf16* pp = &Ps[wave * 16 * LDP];
#pragma unroll
    for (int r = 0; r < 8; ++r) {
      int prow = hilane * 8 + r;
      pp[prow * LDP + mlane]      = (bf16)e0v[r];
      pp[prow * LDP + 16 + mlane] = (bf16)e1v[r];
    }
    v16bf pa = ld_frag_a(pp + mlane * LDP + hilane * 16);

    // ---- O[16x64] += P(16x32) x V(32x64) ----
#pragma unroll
    for (int dt = 0; dt < 4; ++dt) {
      v16bf bv = ld_frag_b(&vtb[(dt * 16 + mlane) * LDV + hilane * 16]);
      o[dt] = WMMA_BF16(pa, bv, o[dt]);
    }
    cur ^= 1;
  }

  // ---- normalize and write bf16 attention output [B*S, E] ----
#pragma unroll
  for (int dt = 0; dt < 4; ++dt) {
    int d = dt * 16 + mlane;
#pragma unroll
    for (int r = 0; r < 8; ++r) {
      int row = q0 + wave * 16 + hilane * 8 + r;
      float v = o[dt][r] / lrow[r];
      attn_out[(rowbase + row) * (size_t)E + h * Dh + d] = (bf16)v;
    }
  }
}

// ---------------------------------------------------------------------------
extern "C" void kernel_launch(void* const* d_in, const int* in_sizes, int n_in,
                              void* d_out, int out_size, void* d_ws, size_t ws_size,
                              hipStream_t stream) {
  (void)in_sizes; (void)n_in; (void)out_size; (void)ws_size;
  const int B = 4, S = 2048, E = 1024;
  const int M = B * S;    // 8192
  const int N3 = 3 * E;   // 3072

  const float* hs  = (const float*)d_in[0];
  const float* caw = (const float*)d_in[1];
  const float* cab = (const float*)d_in[2];
  const float* cpw = (const float*)d_in[3];
  const float* cpb = (const float*)d_in[4];
  float* out = (float*)d_out;

  bf16* qkv  = (bf16*)d_ws;                                          // 8192x3072
  bf16* attn = (bf16*)((char*)d_ws + (size_t)M * N3 * sizeof(bf16)); // 8192x1024

  dim3 g1(N3 / 128, M / 128);
  wmma_gemm_kernel<true, true><<<g1, 256, 0, stream>>>(hs, caw, cab, qkv, M, N3, E);

  dim3 g2(S / 64, B * 16);
  flash_attn_kernel<<<g2, 128, 0, stream>>>(qkv, attn);

  dim3 g3(E / 128, M / 128);
  wmma_gemm_kernel<false, false><<<g3, 256, 0, stream>>>(attn, cpw, cpb, out, M, E, E);
}
